// SimpleBaselineProt_45294725103836
// MI455X (gfx1250) — matrix operand order repaired and checked by reading, not compiled, not run on hardware
//
#include <hip/hip_runtime.h>

// ---------------- problem constants (from reference) ----------------
#define NUM_PROTEINS 19000
#define NUM_DRUGS    4000
#define NUM_CL       16
#define BATCH        4096
#define NUM_EDGES    40000
#define L0           2048     // hidden 0
#define L1           1024     // hidden 1
#define IN_DIM       38032    // 2*(19000+16)
#define ROWS         8192     // 2*BATCH (x1 rows then x2 rows)
#define PCHUNK       6400     // proteins per LDS pass (3 passes cover 19000)

typedef __attribute__((ext_vector_type(16))) _Float16 v16h;
typedef __attribute__((ext_vector_type(8)))  _Float16 v8h;
typedef __attribute__((ext_vector_type(4)))  _Float16 v4h;
typedef __attribute__((ext_vector_type(8)))  float    v8f;

// ------------------------------------------------------------------
// Kernel 1a: dedup edges (reference builds a 0/1 table, duplicates collapse).
// ------------------------------------------------------------------
__global__ void dedup_kernel(const int* __restrict__ dpi_drug,
                             const int* __restrict__ dpi_prot,
                             unsigned* __restrict__ mask,
                             unsigned char* __restrict__ flags) {
    const int e = blockIdx.x * blockDim.x + threadIdx.x;
    if (e >= NUM_EDGES) return;
    const int d = dpi_drug[e];
    const int p = dpi_prot[e];
    const size_t bit = (size_t)d * NUM_PROTEINS + (size_t)p;
    const unsigned old = atomicOr(&mask[bit >> 5], 1u << (bit & 31u));
    flags[e] = ((old >> (bit & 31u)) & 1u) ? 0 : 1;
}

// ------------------------------------------------------------------
// Kernel 1b: per-drug embeddings with coalesced W0 access.
// One block per (W0 row j, protein chunk). The row segment is staged in LDS
// (coalesced global read, W0 touched exactly once), then the per-edge random
// accesses are LDS reads; scatter into E1/E2 is atomic (edges share drugs).
//   E1[d][j] = sum_{p in prot(d)} W0[j][p]
//   E2[d][j] = sum_{p in prot(d)} W0[j][19016 + p]
// ------------------------------------------------------------------
__global__ void embed_rows_kernel(const int* __restrict__ dpi_drug,
                                  const int* __restrict__ dpi_prot,
                                  const unsigned char* __restrict__ flags,
                                  const float* __restrict__ W0,
                                  float* __restrict__ E1,
                                  float* __restrict__ E2) {
    __shared__ float sA[PCHUNK];
    __shared__ float sB[PCHUNK];
    const int j    = blockIdx.x;                       // 0..2047
    const int p0   = blockIdx.y * PCHUNK;
    const int pend = (p0 + PCHUNK < NUM_PROTEINS) ? p0 + PCHUNK : NUM_PROTEINS;
    const size_t rowoff = (size_t)j * IN_DIM;

    for (int p = p0 + threadIdx.x; p < pend; p += blockDim.x) {
        sA[p - p0] = W0[rowoff + p];
        sB[p - p0] = W0[rowoff + (NUM_PROTEINS + NUM_CL) + p];
    }
    __syncthreads();

    for (int e = threadIdx.x; e < NUM_EDGES; e += blockDim.x) {
        if (!flags[e]) continue;
        const int p = dpi_prot[e];
        if (p < p0 || p >= pend) continue;
        const int d = dpi_drug[e];
        atomicAdd(&E1[(size_t)d * L0 + j], sA[p - p0]);
        atomicAdd(&E2[(size_t)d * L0 + j], sB[p - p0]);
    }
}

// ------------------------------------------------------------------
// Kernel 2: per-cell-line constant: CL[cl] = W0[:,19000+cl] + W0[:,38016+cl] + b0
// ------------------------------------------------------------------
__global__ void combine_cl_kernel(const float* __restrict__ W0,
                                  const float* __restrict__ b0,
                                  float* __restrict__ CL) {
    const int cl = blockIdx.x;
    for (int j = threadIdx.x; j < L0; j += blockDim.x) {
        const size_t rowoff = (size_t)j * IN_DIM;
        CL[(size_t)cl * L0 + j] =
            W0[rowoff + NUM_PROTEINS + cl] +
            W0[rowoff + (2 * NUM_PROTEINS + NUM_CL) + cl] + b0[j];
    }
}

// ------------------------------------------------------------------
// Kernel 3: W1 -> f16 (vectorized)
// ------------------------------------------------------------------
__global__ void convert_w1_kernel(const float* __restrict__ W1,
                                  _Float16* __restrict__ W1h) {
    const size_t i4 = (size_t)blockIdx.x * blockDim.x + threadIdx.x;
    if (i4 >= (size_t)L1 * L0 / 4) return;
    const float4 v = ((const float4*)W1)[i4];
    v4h h; h[0] = (_Float16)v.x; h[1] = (_Float16)v.y;
           h[2] = (_Float16)v.z; h[3] = (_Float16)v.w;
    ((v4h*)W1h)[i4] = h;
}

// ------------------------------------------------------------------
// Kernel 4: assemble layer-0 activations (vectorized):
//   row r < 4096 : x1 order; row r >= 4096: x2 order (drugs swapped)
//   H0h[r] = relu(E1[dA] + E2[dB] + CL[cl]) -> f16
// ------------------------------------------------------------------
__global__ void assemble_h0_kernel(const int* __restrict__ drug_pairs,
                                   const int* __restrict__ cell_lines,
                                   const float* __restrict__ E1,
                                   const float* __restrict__ E2,
                                   const float* __restrict__ CL,
                                   _Float16* __restrict__ H0h) {
    const int r = blockIdx.x;
    const int b = r & (BATCH - 1);
    const int swap = r >> 12;
    const int d0 = drug_pairs[2 * b + 0];
    const int d1 = drug_pairs[2 * b + 1];
    const int dA = swap ? d1 : d0;
    const int dB = swap ? d0 : d1;
    const int cl = cell_lines[b];
    const float4* e1 = (const float4*)(E1 + (size_t)dA * L0);
    const float4* e2 = (const float4*)(E2 + (size_t)dB * L0);
    const float4* cc = (const float4*)(CL + (size_t)cl * L0);
    v4h* out = (v4h*)(H0h + (size_t)r * L0);
    for (int j4 = threadIdx.x; j4 < L0 / 4; j4 += blockDim.x) {
        const float4 a = e1[j4], bb = e2[j4], c = cc[j4];
        float x = a.x + bb.x + c.x, y = a.y + bb.y + c.y;
        float z = a.z + bb.z + c.z, w = a.w + bb.w + c.w;
        v4h h;
        h[0] = (_Float16)(x > 0.f ? x : 0.f);
        h[1] = (_Float16)(y > 0.f ? y : 0.f);
        h[2] = (_Float16)(z > 0.f ? z : 0.f);
        h[3] = (_Float16)(w > 0.f ? w : 0.f);
        out[j4] = h;
    }
}

// ------------------------------------------------------------------
// Kernel 5: layer-1 GEMM, WMMA f16->f32, register double-buffered.
//   H1[ROWS x L1] = relu(H0h[ROWS x L0] @ W1h^T + b1)
// One wave computes 32(M) x 64(N): 8 accumulators, 8 WMMAs per 32-K step.
// Next K-step's fragments are loaded before the current step's WMMAs so the
// scheduler overlaps global loads with the matrix pipe. __launch_bounds__
// (256,1) grants the full VGPR budget (~176/lane) so nothing spills.
// Final 64-K chunk is peeled so the steady-state loop has no clamping selects.
// Fragment layouts used here (A/B share one k order) (lane = 16*kh + sub):
//   A 16x32 f16 : one contiguous 16-half chunk at K = 16*kh
//   B 32x16 f16 : one contiguous 16-half chunk at K = 16*kh (B[k][n]=W1[n][k])
//   C/D 16x16 f32: c[rr] is row M = rr + 8*kh, col N = sub
// ------------------------------------------------------------------
#define WMMA_F16(A, B, C) \
    __builtin_amdgcn_wmma_f32_16x16x32_f16(false, (A), false, (B), (short)0, (C), false, false)

__global__ void __launch_bounds__(256, 1)
gemm1_wmma_kernel(const _Float16* __restrict__ H0h,
                  const _Float16* __restrict__ W1h,
                  const float* __restrict__ b1,
                  float* __restrict__ H1) {
    const int wave = (blockIdx.x * blockDim.x + threadIdx.x) >> 5;  // 0..4095
    const int lane = threadIdx.x & 31;
    const int tile_m = (wave >> 4) << 5;     // 256 M-tiles of 32
    const int tile_n = (wave & 15) << 6;     // 16 N-tiles of 64
    const int sub    = lane & 15;
    const int kh     = lane >> 4;

    const _Float16* A0 = H0h + (size_t)(tile_m + sub) * L0 + 16 * kh;
    const _Float16* A1 = A0 + (size_t)16 * L0;
    const _Float16* B0 = W1h + (size_t)(tile_n + sub) * L0 + 16 * kh;

    auto loadA = [](const _Float16* base, int k) -> v16h {
        return *(const v16h*)(base + k);
    };

    v8f acc[8];
#pragma unroll
    for (int i = 0; i < 8; ++i) acc[i] = (v8f){};

    // prologue: fragments for K-step 0
    v16h a0c = loadA(A0, 0);
    v16h a1c = loadA(A1, 0);
    v16h bc0 = *(const v16h*)(B0 + 0 * 16 * L0);
    v16h bc1 = *(const v16h*)(B0 + 1 * 16 * L0);
    v16h bc2 = *(const v16h*)(B0 + 2 * 16 * L0);
    v16h bc3 = *(const v16h*)(B0 + 3 * 16 * L0);

#pragma unroll 2
    for (int k0 = 0; k0 < L0 - 64; k0 += 64) {
        const int k1 = k0 + 32;
        __builtin_prefetch(A0 + k0 + 512, 0, 1);
        __builtin_prefetch(B0 + k0 + 512, 0, 1);

        // loads for K-step k0+32 (independent of current WMMAs)
        v16h a0n = loadA(A0, k1);
        v16h a1n = loadA(A1, k1);
        v16h bn0 = *(const v16h*)(B0 + 0 * 16 * L0 + k1);
        v16h bn1 = *(const v16h*)(B0 + 1 * 16 * L0 + k1);
        v16h bn2 = *(const v16h*)(B0 + 2 * 16 * L0 + k1);
        v16h bn3 = *(const v16h*)(B0 + 3 * 16 * L0 + k1);

        acc[0] = WMMA_F16(a0c, bc0, acc[0]);
        acc[1] = WMMA_F16(a0c, bc1, acc[1]);
        acc[2] = WMMA_F16(a0c, bc2, acc[2]);
        acc[3] = WMMA_F16(a0c, bc3, acc[3]);
        acc[4] = WMMA_F16(a1c, bc0, acc[4]);
        acc[5] = WMMA_F16(a1c, bc1, acc[5]);
        acc[6] = WMMA_F16(a1c, bc2, acc[6]);
        acc[7] = WMMA_F16(a1c, bc3, acc[7]);

        // loads for K-step k0+64 (always in range: loop stops 64 early)
        const int k2 = k0 + 64;
        a0c = loadA(A0, k2);
        a1c = loadA(A1, k2);
        bc0 = *(const v16h*)(B0 + 0 * 16 * L0 + k2);
        bc1 = *(const v16h*)(B0 + 1 * 16 * L0 + k2);
        bc2 = *(const v16h*)(B0 + 2 * 16 * L0 + k2);
        bc3 = *(const v16h*)(B0 + 3 * 16 * L0 + k2);

        acc[0] = WMMA_F16(a0n, bn0, acc[0]);
        acc[1] = WMMA_F16(a0n, bn1, acc[1]);
        acc[2] = WMMA_F16(a0n, bn2, acc[2]);
        acc[3] = WMMA_F16(a0n, bn3, acc[3]);
        acc[4] = WMMA_F16(a1n, bn0, acc[4]);
        acc[5] = WMMA_F16(a1n, bn1, acc[5]);
        acc[6] = WMMA_F16(a1n, bn2, acc[6]);
        acc[7] = WMMA_F16(a1n, bn3, acc[7]);
    }

    // peeled tail: K-steps L0-64 and L0-32 (cur already holds L0-64)
    {
        const int k1 = L0 - 32;
        v16h a0n = loadA(A0, k1);
        v16h a1n = loadA(A1, k1);
        v16h bn0 = *(const v16h*)(B0 + 0 * 16 * L0 + k1);
        v16h bn1 = *(const v16h*)(B0 + 1 * 16 * L0 + k1);
        v16h bn2 = *(const v16h*)(B0 + 2 * 16 * L0 + k1);
        v16h bn3 = *(const v16h*)(B0 + 3 * 16 * L0 + k1);

        acc[0] = WMMA_F16(a0c, bc0, acc[0]);
        acc[1] = WMMA_F16(a0c, bc1, acc[1]);
        acc[2] = WMMA_F16(a0c, bc2, acc[2]);
        acc[3] = WMMA_F16(a0c, bc3, acc[3]);
        acc[4] = WMMA_F16(a1c, bc0, acc[4]);
        acc[5] = WMMA_F16(a1c, bc1, acc[5]);
        acc[6] = WMMA_F16(a1c, bc2, acc[6]);
        acc[7] = WMMA_F16(a1c, bc3, acc[7]);

        acc[0] = WMMA_F16(a0n, bn0, acc[0]);
        acc[1] = WMMA_F16(a0n, bn1, acc[1]);
        acc[2] = WMMA_F16(a0n, bn2, acc[2]);
        acc[3] = WMMA_F16(a0n, bn3, acc[3]);
        acc[4] = WMMA_F16(a1n, bn0, acc[4]);
        acc[5] = WMMA_F16(a1n, bn1, acc[5]);
        acc[6] = WMMA_F16(a1n, bn2, acc[6]);
        acc[7] = WMMA_F16(a1n, bn3, acc[7]);
    }

    // epilogue: bias + relu, store per f32 C/D layout
#pragma unroll
    for (int mi = 0; mi < 2; ++mi) {
#pragma unroll
        for (int ns = 0; ns < 4; ++ns) {
            const int col = tile_n + ns * 16 + sub;
            const float bias = b1[col];
            const v8f c = acc[mi * 4 + ns];
#pragma unroll
            for (int rr = 0; rr < 8; ++rr) {
                float v = c[rr] + bias;
                v = v > 0.0f ? v : 0.0f;
                H1[(size_t)(tile_m + mi * 16 + rr + 8 * kh) * L1 + col] = v;
            }
        }
    }
}

// ------------------------------------------------------------------
// Kernel 6: layer 2: out[r] = dot(H1[r], W2) + b2   (one wave per row)
// ------------------------------------------------------------------
__global__ void layer2_kernel(const float* __restrict__ H1,
                              const float* __restrict__ W2,
                              const float* __restrict__ b2,
                              float* __restrict__ out) {
    const int wave = (blockIdx.x * blockDim.x + threadIdx.x) >> 5;  // 0..8191
    const int lane = threadIdx.x & 31;
    const float4* h = (const float4*)(H1 + (size_t)wave * L1);
    const float4* w = (const float4*)W2;
    float s = 0.0f;
#pragma unroll
    for (int t = 0; t < 8; ++t) {
        const int i = lane + 32 * t;
        const float4 a = h[i];
        const float4 bb = w[i];
        s += a.x * bb.x + a.y * bb.y + a.z * bb.z + a.w * bb.w;
    }
#pragma unroll
    for (int off = 16; off > 0; off >>= 1) s += __shfl_down(s, off, 32);
    if (lane == 0) out[wave] = s + b2[0];
}

// ------------------------------------------------------------------
// host-side orchestration
// ------------------------------------------------------------------
static inline size_t align_up(size_t x, size_t a) { return (x + a - 1) & ~(a - 1); }

extern "C" void kernel_launch(void* const* d_in, const int* in_sizes, int n_in,
                              void* d_out, int out_size, void* d_ws, size_t ws_size,
                              hipStream_t stream) {
    const int*   drug_pairs = (const int*)d_in[0];
    const int*   cell_lines = (const int*)d_in[1];
    const int*   dpi_drug   = (const int*)d_in[2];
    const int*   dpi_prot   = (const int*)d_in[3];
    const float* W0         = (const float*)d_in[4];
    const float* b0         = (const float*)d_in[5];
    const float* W1         = (const float*)d_in[6];
    const float* b1         = (const float*)d_in[7];
    const float* W2         = (const float*)d_in[8];
    const float* b2         = (const float*)d_in[9];
    float*       out        = (float*)d_out;

    // workspace layout (mask, E1, E2 contiguous so one memset zeros them all)
    const size_t maskWords = ((size_t)NUM_DRUGS * NUM_PROTEINS + 31) / 32; // 2,375,000
    size_t off = 0;
    const size_t off_mask  = off; off = align_up(off + maskWords * 4, 256);
    const size_t off_E1    = off; off = align_up(off + (size_t)NUM_DRUGS * L0 * 4, 256);
    const size_t off_E2    = off; off = align_up(off + (size_t)NUM_DRUGS * L0 * 4, 256);
    const size_t zero_bytes = off;                  // mask + E1 + E2
    const size_t off_flags = off; off = align_up(off + NUM_EDGES, 256);
    const size_t off_CL    = off; off = align_up(off + (size_t)NUM_CL * L0 * 4, 256);
    const size_t off_H0h   = off; off = align_up(off + (size_t)ROWS * L0 * 2, 256);
    const size_t off_W1h   = off; off = align_up(off + (size_t)L1 * L0 * 2, 256);
    const size_t off_H1    = off; off = align_up(off + (size_t)ROWS * L1 * 4, 256);
    (void)ws_size; (void)in_sizes; (void)n_in; (void)out_size;

    char* ws = (char*)d_ws;
    unsigned*      mask  = (unsigned*)(ws + off_mask);
    float*         E1    = (float*)(ws + off_E1);
    float*         E2    = (float*)(ws + off_E2);
    unsigned char* flags = (unsigned char*)(ws + off_flags);
    float*         CL    = (float*)(ws + off_CL);
    _Float16*      H0h   = (_Float16*)(ws + off_H0h);
    _Float16*      W1h   = (_Float16*)(ws + off_W1h);
    float*         H1    = (float*)(ws + off_H1);

    // zero accumulation buffers (graph-capture safe)
    hipMemsetAsync(ws, 0, zero_bytes, stream);

    dedup_kernel<<<(NUM_EDGES + 255) / 256, 256, 0, stream>>>(dpi_drug, dpi_prot, mask, flags);

    {   // coalesced W0 pass: 2048 rows x 3 protein chunks
        dim3 grid(L0, (NUM_PROTEINS + PCHUNK - 1) / PCHUNK);
        embed_rows_kernel<<<grid, 256, 0, stream>>>(dpi_drug, dpi_prot, flags, W0, E1, E2);
    }

    combine_cl_kernel<<<NUM_CL, 256, 0, stream>>>(W0, b0, CL);
    convert_w1_kernel<<<(L1 * L0 / 4 + 255) / 256, 256, 0, stream>>>(W1, W1h);
    assemble_h0_kernel<<<ROWS, 256, 0, stream>>>(drug_pairs, cell_lines, E1, E2, CL, H0h);

    // 4096 waves, one per 32x64 tile -> 512 blocks of 8 waves
    gemm1_wmma_kernel<<<(ROWS / 32) * (L1 / 64) / 8, 256, 0, stream>>>(H0h, W1h, b1, H1);

    // 8192 waves, one per output row
    layer2_kernel<<<ROWS / 8, 256, 0, stream>>>(H1, W2, b2, out);
}
